// MoEDist_5566277615661
// MI455X (gfx1250) — matrix operand order repaired and checked by reading, not compiled, not run on hardware
//
#include <hip/hip_runtime.h>
#include <hip/hip_bf16.h>

typedef __attribute__((ext_vector_type(16))) _Float16     v16h;
typedef __attribute__((ext_vector_type(8)))  _Float16     v8h;
typedef __attribute__((ext_vector_type(8)))  float        v8f;
typedef __attribute__((ext_vector_type(4)))  unsigned int v4u;
typedef __attribute__((ext_vector_type(8)))  int          v8i_;
typedef __attribute__((ext_vector_type(4)))  int          v4i_;

#define B_  8
#define S_  4096
#define M_  1024
#define H_  4096
#define E_  8
#define L_  (B_*S_)      /* 32768 tokens */
#define C_  (L_/E_)      /* 4096 capacity */

#define TM 16            /* slot rows per workgroup */
#define XA_STRIDE 1024   /* TDM writes rows packed: no pad */
#define HS_STRIDE 4104   /* 4096 + 8 f16 pad (16B -> 4-bank shift per row) */
#define HS_BYTES  ((size_t)TM * HS_STRIDE * sizeof(_Float16))
#define SMEM_BYTES (HS_BYTES + (size_t)TM * XA_STRIDE * sizeof(_Float16))

// ---------------------------------------------------------------------------
// 1) Deterministic routing: exact cumsum-rank per expert + capacity drop.
// ---------------------------------------------------------------------------
__global__ __launch_bounds__(1024) void route_scan(
    const int* __restrict__ eid,
    int* __restrict__ slot_of_token,   // [L] : e*C+pos, or -1 if dropped
    int* __restrict__ token_of_slot)   // [E*C] : token index, or -1 if empty
{
    __shared__ int scan[1024][E_];     // 32 KB
    const int t    = threadIdx.x;
    const int base = t * 32;

    #pragma unroll
    for (int j = 0; j < 32; ++j) token_of_slot[base + j] = -1;

    int cnt[E_];
    #pragma unroll
    for (int e = 0; e < E_; ++e) cnt[e] = 0;
    for (int j = 0; j < 32; ++j) cnt[eid[base + j]]++;

    #pragma unroll
    for (int e = 0; e < E_; ++e) scan[t][e] = cnt[e];
    __syncthreads();

    for (int off = 1; off < 1024; off <<= 1) {
        int v[E_];
        if (t >= off) {
            #pragma unroll
            for (int e = 0; e < E_; ++e) v[e] = scan[t - off][e];
        }
        __syncthreads();
        if (t >= off) {
            #pragma unroll
            for (int e = 0; e < E_; ++e) scan[t][e] += v[e];
        }
        __syncthreads();
    }

    int pos[E_];
    #pragma unroll
    for (int e = 0; e < E_; ++e) pos[e] = scan[t][e] - cnt[e];  // exclusive

    for (int j = 0; j < 32; ++j) {
        const int l  = base + j;
        const int e  = eid[l];
        const int p  = pos[e]++;
        const int sl = (p < C_) ? (e * C_ + p) : -1;
        slot_of_token[l] = sl;
        if (sl >= 0) token_of_slot[sl] = l;
    }
}

// ---------------------------------------------------------------------------
// 2) Tiled transpose + f32->f16 convert (per expert): in[r][c] -> out[c][r]
// ---------------------------------------------------------------------------
__global__ __launch_bounds__(256) void transpose_cvt_f16(
    const float* __restrict__ in, _Float16* __restrict__ outp,
    int rows, int cols)
{
    __shared__ _Float16 tile[32][33];
    const int e = blockIdx.z;
    const float* ine  = in   + (size_t)e * rows * cols;
    _Float16*    oute = outp + (size_t)e * rows * cols;
    const int c0 = blockIdx.x * 32, r0 = blockIdx.y * 32;
    const int tx = threadIdx.x, ty = threadIdx.y;        // 32 x 8

    #pragma unroll
    for (int i = 0; i < 32; i += 8)
        tile[ty + i][tx] = (_Float16)ine[(size_t)(r0 + ty + i) * cols + c0 + tx];
    __syncthreads();
    #pragma unroll
    for (int i = 0; i < 32; i += 8)
        oute[(size_t)(c0 + ty + i) * rows + r0 + tx] = tile[tx][ty + i];
}

// ---------------------------------------------------------------------------
// 3) X f32 -> f16, same layout (A operand is K-contiguous already).
// ---------------------------------------------------------------------------
__global__ __launch_bounds__(256) void cvt_x_f16(
    const float* __restrict__ X, _Float16* __restrict__ X16)
{
    const size_t i0 = ((size_t)blockIdx.x * 256 + threadIdx.x) * 8;
    #pragma unroll
    for (int j = 0; j < 8; ++j) X16[i0 + j] = (_Float16)X[i0 + j];
}

// ---------------------------------------------------------------------------
// 4) Dropped tokens produce zero rows in the output.
// ---------------------------------------------------------------------------
__global__ __launch_bounds__(256) void zero_dropped(
    const int* __restrict__ slot_of_token, float* __restrict__ out)
{
    const int l = blockIdx.x;
    if (slot_of_token[l] >= 0) return;
    float* row = out + (size_t)l * M_;
    for (int c = threadIdx.x; c < M_; c += 256) row[c] = 0.0f;
}

// ---------------------------------------------------------------------------
// 5) Fused per-expert FFN. One block = (expert e, 16 slot rows), 8 wave32.
//    Gather of the 16 token rows into LDS is done by the Tensor Data Mover
//    in gather mode (16-bit row indices, tile 16 x 1024 f16), tracked by
//    TENSORcnt. Each wave computes two adjacent 16x16 n-tiles per A-fragment.
// ---------------------------------------------------------------------------
__device__ __forceinline__ v16h load_a_frag(const _Float16* arow, int koff) {
    v8h lo = *(const v8h*)(arow + koff);
    v8h hi = *(const v8h*)(arow + koff + 8);
    v16h a;
    #pragma unroll
    for (int i = 0; i < 8; ++i) { a[i] = lo[i]; a[i + 8] = hi[i]; }
    return a;
}

__device__ __forceinline__ unsigned rl_u32(unsigned x) {
    return (unsigned)__builtin_amdgcn_readfirstlane((int)x);
}

__global__ __launch_bounds__(256) void moe_ffn(
    const _Float16* __restrict__ X16,           // [L][M] f16
    const _Float16* __restrict__ W1T,           // [E][H][M] f16 (transposed)
    const float*    __restrict__ b1,            // [E][H]
    const _Float16* __restrict__ W2T,           // [E][M][H] f16 (transposed)
    const float*    __restrict__ b2,            // [E][M]
    const int*      __restrict__ token_of_slot, // [E*C]
    float*          __restrict__ out)           // [L][M] f32
{
    extern __shared__ __align__(16) char smem[];
    _Float16* Hs = (_Float16*)smem;                      // [TM][HS_STRIDE]
    _Float16* Xa = (_Float16*)(smem + HS_BYTES);         // [TM][XA_STRIDE]

    const int e    = blockIdx.y;
    const int i0   = blockIdx.x * TM;
    const int tid  = threadIdx.x;
    const int wave = tid >> 5;
    const int lane = tid & 31;
    const int ln16 = lane & 15;
    const int kh   = lane >> 4;

#if __has_builtin(__builtin_amdgcn_tensor_load_to_lds)
    // --- TDM gather: 16 rows of X16 (indexed by token id) -> Xa ---
    {
        unsigned idx[TM];
        #pragma unroll
        for (int r = 0; r < TM; ++r) {
            int t = token_of_slot[e * C_ + i0 + r];
            idx[r] = rl_u32((unsigned)(t < 0 ? 0 : t));   // empty rows: harmless row 0
        }
        const unsigned lds_off =
            rl_u32((unsigned)__builtin_amdgcn_groupstaticsize() + (unsigned)HS_BYTES);
        const unsigned long long ga = (unsigned long long)(uintptr_t)X16;

        v4u g0;
        g0[0] = rl_u32(1u | (1u << 31));                  // count=1, gather_mode=1, 16-bit idx
        g0[1] = lds_off;                                  // lds_addr
        g0[2] = rl_u32((unsigned)(ga & 0xFFFFFFFFu));     // global_addr[31:0]
        g0[3] = rl_u32((unsigned)((ga >> 32) & 0x1FFFFFFu) | (2u << 30)); // addr[56:32], type=2

        v8i_ g1;
        g1[0] = (int)rl_u32(1u << 16);                    // data_size=1 (2 bytes)
        g1[1] = (int)rl_u32(((unsigned)M_ & 0xFFFFu) << 16);              // tensor_dim0 lo16
        g1[2] = (int)rl_u32((((unsigned)L_ & 0xFFFFu) << 16) | ((unsigned)M_ >> 16)); // dim0 hi16, dim1 lo16
        g1[3] = (int)rl_u32(((unsigned)M_ << 16) | ((unsigned)L_ >> 16)); // dim1 hi16, tile_dim0=1024
        g1[4] = (int)rl_u32((unsigned)TM);                // tile_dim1 = #valid indices = 16
        g1[5] = (int)rl_u32((unsigned)M_);                // tensor_dim0_stride lo32
        g1[6] = 0;
        g1[7] = 0;

        v4i_ g2, g3;
        #pragma unroll
        for (int i = 0; i < 4; ++i)
            g2[i] = (int)rl_u32(idx[2*i] | (idx[2*i + 1] << 16));
        #pragma unroll
        for (int i = 0; i < 4; ++i)
            g3[i] = (int)rl_u32(idx[8 + 2*i] | (idx[8 + 2*i + 1] << 16));

        v8i_ g4 = {};                                     // reserved/extra group (6-arg form)

        if (wave == 0) {
            __builtin_amdgcn_tensor_load_to_lds(g0, g1, g2, g3, g4, 0);
            __builtin_amdgcn_s_wait_tensorcnt(0);
        }
    }
    __syncthreads();
#else
    // --- fallback: manual gather (f16 copy) ---
    {
        const int r  = tid >> 4;
        const int cl = tid & 15;
        const int t  = token_of_slot[e * C_ + i0 + r];
        const _Float16* src = X16 + (size_t)(t < 0 ? 0 : t) * M_;
        for (int c = cl; c < M_; c += 16)
            Xa[r * XA_STRIDE + c] = (t < 0) ? (_Float16)0.0f : src[c];
    }
    __syncthreads();
#endif

    const _Float16* W1e = W1T + (size_t)e * H_ * M_;
    const float*    b1e = b1 + e * H_;

    // --- stage A: Hs[16][H] = relu(Xa @ W1 + b1), K = M = 1024 ---
    for (int p = wave; p < H_ / 32; p += 8) {
        const int nA = p * 32 + ln16;
        const int nB = nA + 16;
        v8f acc0 = {}, acc1 = {};
        const _Float16* bcol0 = W1e + (size_t)nA * M_ + kh * 16;
        const _Float16* bcol1 = W1e + (size_t)nB * M_ + kh * 16;
        const _Float16* arow  = Xa + ln16 * XA_STRIDE + kh * 16;
        #pragma unroll 4
        for (int ks = 0; ks < M_ / 32; ++ks) {
            v16h a  = load_a_frag(arow, ks * 32);
            v16h b0 = *(const v16h*)(bcol0 + ks * 32);
            v16h b1v = *(const v16h*)(bcol1 + ks * 32);
            acc0 = __builtin_amdgcn_wmma_f32_16x16x32_f16(
                false, a, false, b0, (short)0, acc0, false, false);
            acc1 = __builtin_amdgcn_wmma_f32_16x16x32_f16(
                false, a, false, b1v, (short)0, acc1, false, false);
        }
        const float biasA = b1e[nA];
        const float biasB = b1e[nB];
        #pragma unroll
        for (int v = 0; v < 8; ++v) {
            float h0 = acc0[v] + biasA; h0 = h0 > 0.0f ? h0 : 0.0f;
            float h1 = acc1[v] + biasB; h1 = h1 > 0.0f ? h1 : 0.0f;
            Hs[(v + 8 * kh) * HS_STRIDE + nA] = (_Float16)h0;  // D f32 layout: (M=v+8*kh, N)
            Hs[(v + 8 * kh) * HS_STRIDE + nB] = (_Float16)h1;
        }
    }
    __syncthreads();

    // per-lane output-row token indices (rows v + 8*kh), reused across n-tiles
    int trow[8];
    #pragma unroll
    for (int v = 0; v < 8; ++v)
        trow[v] = token_of_slot[e * C_ + i0 + v + 8 * kh];

    const _Float16* W2e = W2T + (size_t)e * M_ * H_;
    const float*    b2e = b2 + e * M_;

    // --- stage B: out rows = Hs @ W2 + b2, K = H = 4096, fused combine ---
    for (int p = wave; p < M_ / 32; p += 8) {
        const int nA = p * 32 + ln16;
        const int nB = nA + 16;
        v8f acc0 = {}, acc1 = {};
        const _Float16* bcol0 = W2e + (size_t)nA * H_ + kh * 16;
        const _Float16* bcol1 = W2e + (size_t)nB * H_ + kh * 16;
        const _Float16* arow  = Hs + ln16 * HS_STRIDE + kh * 16;
        #pragma unroll 4
        for (int ks = 0; ks < H_ / 32; ++ks) {
            v16h a  = load_a_frag(arow, ks * 32);
            v16h b0 = *(const v16h*)(bcol0 + ks * 32);
            v16h b1v = *(const v16h*)(bcol1 + ks * 32);
            acc0 = __builtin_amdgcn_wmma_f32_16x16x32_f16(
                false, a, false, b0, (short)0, acc0, false, false);
            acc1 = __builtin_amdgcn_wmma_f32_16x16x32_f16(
                false, a, false, b1v, (short)0, acc1, false, false);
        }
        const float biasA = b2e[nA];
        const float biasB = b2e[nB];
        #pragma unroll
        for (int v = 0; v < 8; ++v) {
            if (trow[v] >= 0) {
                float* orow = out + (size_t)trow[v] * M_;
                orow[nA] = acc0[v] + biasA;
                orow[nB] = acc1[v] + biasB;
            }
        }
    }
}

// ---------------------------------------------------------------------------
// launch
// ---------------------------------------------------------------------------
extern "C" void kernel_launch(void* const* d_in, const int* in_sizes, int n_in,
                              void* d_out, int out_size, void* d_ws, size_t ws_size,
                              hipStream_t stream)
{
    const float* X   = (const float*)d_in[0];   // [B,S,M]
    const int*   eid = (const int*)  d_in[1];   // [B,S]
    const float* W1  = (const float*)d_in[2];   // [E,M,H]
    const float* b1  = (const float*)d_in[3];   // [E,H]
    const float* W2  = (const float*)d_in[4];   // [E,H,M]
    const float* b2  = (const float*)d_in[5];   // [E,M]
    float*       out = (float*)d_out;           // [B,S,M]

    char* ws = (char*)d_ws;
    _Float16* X16 = (_Float16*)ws;  ws += (size_t)L_ * M_ * sizeof(_Float16);
    _Float16* W1T = (_Float16*)ws;  ws += (size_t)E_ * H_ * M_ * sizeof(_Float16);
    _Float16* W2T = (_Float16*)ws;  ws += (size_t)E_ * M_ * H_ * sizeof(_Float16);
    int* token_of_slot = (int*)ws;  ws += (size_t)L_ * sizeof(int);
    int* slot_of_token = (int*)ws;

    route_scan<<<1, 1024, 0, stream>>>(eid, slot_of_token, token_of_slot);

    cvt_x_f16<<<(int)((size_t)L_ * M_ / (256 * 8)), 256, 0, stream>>>(X, X16);

    dim3 tb(32, 8);
    // W1 [E][M rows][H cols] -> W1T [E][H][M]
    transpose_cvt_f16<<<dim3(H_ / 32, M_ / 32, E_), tb, 0, stream>>>(W1, W1T, M_, H_);
    // W2 [E][H rows][M cols] -> W2T [E][M][H]
    transpose_cvt_f16<<<dim3(M_ / 32, H_ / 32, E_), tb, 0, stream>>>(W2, W2T, H_, M_);

    zero_dropped<<<L_, 256, 0, stream>>>(slot_of_token, out);

    (void)hipFuncSetAttribute((const void*)moe_ffn,
                              hipFuncAttributeMaxDynamicSharedMemorySize,
                              (int)SMEM_BYTES);
    moe_ffn<<<dim3(C_ / TM, E_), 256, SMEM_BYTES, stream>>>(
        X16, W1T, b1, W2T, b2, token_of_slot, out);
}
